// Qwen3MoeFusedSparseMoeBlock_31585189495029
// MI455X (gfx1250) — matrix operand order repaired and checked by reading, not compiled, not run on hardware
//
#include <hip/hip_runtime.h>
#include <hip/hip_bf16.h>

// ---------------------------------------------------------------------------
// Qwen3-MoE fused sparse MoE block for gfx1250 (MI455X).
// bf16 WMMA (v_wmma_f32_16x16x32_bf16) with fp32 accumulation.
// 64x64 block tiles, ping-pong LDS double buffering, 1 barrier / K-chunk.
// ---------------------------------------------------------------------------

#define NE 8          // experts
#define TT 2048       // tokens = B*S
#define DD 2048       // hidden
#define FF 1024       // intermediate
#define TCAP 2048     // max tokens per expert

typedef __attribute__((ext_vector_type(16))) __bf16 v16bf;
typedef __attribute__((ext_vector_type(8)))  __bf16 v8bf;
typedef __attribute__((ext_vector_type(8)))  float  v8f;

#define LDS_STRIDE 40   // bf16 elems/row (32 data + 8 pad): conflict-free frag reads

// ---------------------------------------------------------------------------
// Kernel 0: zero the dense output region (atomics accumulate into it later)
// ---------------------------------------------------------------------------
__global__ void k_zero(float* __restrict__ out, int n) {
  int i = (blockIdx.x * blockDim.x + threadIdx.x) * 4;
  if (i < n) {
    float4 z = make_float4(0.f, 0.f, 0.f, 0.f);
    *(float4*)(out + i) = z;
  }
}

// ---------------------------------------------------------------------------
// Kernel 1: router. One wave per token. Writes router_logits (pre-softmax,
// fp32) and per-token (expert id, renormalized top-2 weight).
// ---------------------------------------------------------------------------
__global__ void k_router(const float* __restrict__ x,
                         const float* __restrict__ gw,
                         float* __restrict__ logits_out,
                         int* __restrict__ sel,
                         float* __restrict__ selw) {
  int wave = threadIdx.x >> 5;
  int lane = threadIdx.x & 31;
  int t = blockIdx.x * 8 + wave;
  if (t >= TT) return;

  const float* xr = x + (size_t)t * DD;
  float acc[NE];
#pragma unroll
  for (int e = 0; e < NE; ++e) acc[e] = 0.f;

  for (int it = 0; it < DD / 128; ++it) {
    float4 xv = *(const float4*)(xr + it * 128 + lane * 4);
#pragma unroll
    for (int e = 0; e < NE; ++e) {
      float4 gv = *(const float4*)(gw + (size_t)e * DD + it * 128 + lane * 4);
      acc[e] += xv.x * gv.x + xv.y * gv.y + xv.z * gv.z + xv.w * gv.w;
    }
  }
#pragma unroll
  for (int e = 0; e < NE; ++e) {
#pragma unroll
    for (int m = 16; m >= 1; m >>= 1) acc[e] += __shfl_xor(acc[e], m, 32);
  }

  if (lane == 0) {
#pragma unroll
    for (int e = 0; e < NE; ++e) logits_out[t * NE + e] = acc[e];
    // top-2 (ties -> lowest index, matching jax top_k)
    int i1 = 0; float b1 = acc[0];
#pragma unroll
    for (int e = 1; e < NE; ++e) if (acc[e] > b1) { b1 = acc[e]; i1 = e; }
    int i2 = -1; float b2 = -3.4e38f;
#pragma unroll
    for (int e = 0; e < NE; ++e)
      if (e != i1 && acc[e] > b2) { b2 = acc[e]; i2 = e; }
    // softmax probs of top-2, renormalized over the pair (denominators cancel)
    float eb = __expf(b2 - b1);
    float inv = __builtin_amdgcn_rcpf(1.f + eb);
    sel[t * 2 + 0] = i1;  selw[t * 2 + 0] = inv;
    sel[t * 2 + 1] = i2;  selw[t * 2 + 1] = eb * inv;
  }
}

// ---------------------------------------------------------------------------
// Kernel 2: deterministic per-expert token-list build (ballot compaction).
// One wave per expert.
// ---------------------------------------------------------------------------
__global__ void k_build(const int* __restrict__ sel,
                        const float* __restrict__ selw,
                        int* __restrict__ counts,
                        int* __restrict__ tokens,
                        float* __restrict__ scales) {
  int e = blockIdx.x;
  int lane = threadIdx.x;
  int cnt = 0;
  for (int t0 = 0; t0 < TT; t0 += 32) {
    int t = t0 + lane;
    int s0 = sel[t * 2], s1 = sel[t * 2 + 1];
    bool pick = (s0 == e) || (s1 == e);
    float w = (s0 == e) ? selw[t * 2] : selw[t * 2 + 1];
    unsigned m32 = (unsigned)__ballot(pick);
    int pos = __popc(m32 & ((1u << lane) - 1u));
    if (pick) {
      tokens[e * TCAP + cnt + pos] = t;
      scales[e * TCAP + cnt + pos] = w;
    }
    cnt += __popc(m32);
  }
  if (lane == 0) counts[e] = cnt;
}

// ---------------------------------------------------------------------------
// Fragment helpers (CDNA5 16-bit WMMA layouts):
//   A: per-lane K-chunks {0-7,16-23} (lanes 0-15) / {8-15,24-31} (lanes 16-31)
//   B: per-lane K-chunks {0-7,16-23} (lanes 0-15) / {8-15,24-31} (lanes 16-31)
// ---------------------------------------------------------------------------
__device__ __forceinline__ v16bf load_a_frag(const __bf16* base, int row, int lh) {
  const __bf16* p = base + row * LDS_STRIDE;
  v8bf lo = *(const v8bf*)(p + 8 * lh);
  v8bf hi = *(const v8bf*)(p + 16 + 8 * lh);
  return __builtin_shufflevector(lo, hi, 0, 1, 2, 3, 4, 5, 6, 7,
                                 8, 9, 10, 11, 12, 13, 14, 15);
}
__device__ __forceinline__ v16bf load_b_frag(const __bf16* base, int row, int lh) {
  const __bf16* p = base + row * LDS_STRIDE;
  v8bf lo = *(const v8bf*)(p + 8 * lh);
  v8bf hi = *(const v8bf*)(p + 16 + 8 * lh);
  return __builtin_shufflevector(lo, hi, 0, 1, 2, 3, 4, 5, 6, 7,
                                 8, 9, 10, 11, 12, 13, 14, 15);
}
__device__ __forceinline__ v8bf cvt8(float4 a, float4 b) {
  v8bf r = { (__bf16)a.x, (__bf16)a.y, (__bf16)a.z, (__bf16)a.w,
             (__bf16)b.x, (__bf16)b.y, (__bf16)b.z, (__bf16)b.w };
  return r;
}

// ---------------------------------------------------------------------------
// Kernel 3: fused gate/up projections + SiLU.
// Block = 256 threads (8 waves, 2x4 wave grid), tile = 64 tokens x 64 F cols,
// K-loop over D in chunks of 32, double-buffered LDS, 1 barrier per chunk.
// grid = (F/64, T/64, E)
// ---------------------------------------------------------------------------
__global__ void __launch_bounds__(256)
k_gateup(const float* __restrict__ x,
         const float* __restrict__ wg,
         const float* __restrict__ wu,
         const int* __restrict__ counts,
         const int* __restrict__ tokens,
         __bf16* __restrict__ hbuf) {
  const int e = blockIdx.z;
  const int cnt = counts[e];
  const int m0 = blockIdx.y * 64;
  if (m0 >= cnt) return;
  const int f0 = blockIdx.x * 64;

  __shared__ __bf16 lA[2][64 * LDS_STRIDE];
  __shared__ __bf16 lBg[2][64 * LDS_STRIDE];
  __shared__ __bf16 lBu[2][64 * LDS_STRIDE];

  const int tid = threadIdx.x;
  const int wv = tid >> 5, ln = tid & 31;
  const int mi = wv >> 2, ni = wv & 3;        // waves: 2 (M) x 4 (N)
  const int lr = ln & 15, lh = ln >> 4;

  // staging map: 64 rows x 32 k; each thread handles 8 consecutive elements
  const int srow = tid >> 2;
  const int skoff = (tid & 3) * 8;
  const int sofs = srow * LDS_STRIDE + skoff;

  int aslot = m0 + srow;
  int atok = tokens[e * TCAP + ((aslot < cnt) ? aslot : m0)];
  const float* aP = x + (size_t)atok * DD + skoff;
  const float* gP = wg + ((size_t)e * FF + f0 + srow) * DD + skoff;
  const float* uP = wu + ((size_t)e * FF + f0 + srow) * DD + skoff;

  v8f accG0 = {}, accG1 = {}, accU0 = {}, accU1 = {};

  // preload chunk 0
  {
    float4 a0 = *(const float4*)(aP),     a1 = *(const float4*)(aP + 4);
    float4 g0 = *(const float4*)(gP),     g1 = *(const float4*)(gP + 4);
    float4 u0 = *(const float4*)(uP),     u1 = *(const float4*)(uP + 4);
    *(v8bf*)(&lA[0][sofs])  = cvt8(a0, a1);
    *(v8bf*)(&lBg[0][sofs]) = cvt8(g0, g1);
    *(v8bf*)(&lBu[0][sofs]) = cvt8(u0, u1);
  }
  __syncthreads();

  int p = 0;
  for (int k0 = 0; k0 < DD; k0 += 32) {
    const bool more = (k0 + 32) < DD;
    float4 na0, na1, ng0, ng1, nu0, nu1;
    if (more) {
      na0 = *(const float4*)(aP + k0 + 32);  na1 = *(const float4*)(aP + k0 + 36);
      ng0 = *(const float4*)(gP + k0 + 32);  ng1 = *(const float4*)(gP + k0 + 36);
      nu0 = *(const float4*)(uP + k0 + 32);  nu1 = *(const float4*)(uP + k0 + 36);
    }

    v16bf aF0 = load_a_frag(lA[p], mi * 32 + lr, lh);
    v16bf aF1 = load_a_frag(lA[p], mi * 32 + 16 + lr, lh);
    v16bf bG  = load_b_frag(lBg[p], ni * 16 + lr, lh);
    v16bf bU  = load_b_frag(lBu[p], ni * 16 + lr, lh);

    accG0 = __builtin_amdgcn_wmma_f32_16x16x32_bf16(false, aF0, false, bG,
                                                    (short)0, accG0, false, false);
    accG1 = __builtin_amdgcn_wmma_f32_16x16x32_bf16(false, aF1, false, bG,
                                                    (short)0, accG1, false, false);
    accU0 = __builtin_amdgcn_wmma_f32_16x16x32_bf16(false, aF0, false, bU,
                                                    (short)0, accU0, false, false);
    accU1 = __builtin_amdgcn_wmma_f32_16x16x32_bf16(false, aF1, false, bU,
                                                    (short)0, accU1, false, false);

    if (more) {
      *(v8bf*)(&lA[1 - p][sofs])  = cvt8(na0, na1);
      *(v8bf*)(&lBg[1 - p][sofs]) = cvt8(ng0, ng1);
      *(v8bf*)(&lBu[1 - p][sofs]) = cvt8(nu0, nu1);
    }
    __syncthreads();
    p ^= 1;
  }

  // epilogue: h = silu(g) * u = g * rcp(1+exp(-g)) * u, store bf16
  const int col = f0 + ni * 16 + lr;
#pragma unroll
  for (int sub = 0; sub < 2; ++sub) {
    const v8f& g8 = sub ? accG1 : accG0;
    const v8f& u8 = sub ? accU1 : accU0;
#pragma unroll
    for (int r = 0; r < 8; ++r) {
      int slot = m0 + mi * 32 + sub * 16 + r + 8 * lh;
      if (slot < cnt) {
        float g = g8[r];
        float u = u8[r];
        float h = g * __builtin_amdgcn_rcpf(1.f + __expf(-g)) * u;
        hbuf[((size_t)e * TCAP + slot) * FF + col] = (__bf16)h;
      }
    }
  }
}

// ---------------------------------------------------------------------------
// Kernel 4: down projection + scaled accumulate into out.
// tile = 64 slots x 64 D cols, K-loop over F in chunks of 32, double-buffered.
// grid = (D/64, T/64, E)
// ---------------------------------------------------------------------------
__global__ void __launch_bounds__(256)
k_down(const __bf16* __restrict__ hbuf,
       const float* __restrict__ wd,
       const int* __restrict__ counts,
       const int* __restrict__ tokens,
       const float* __restrict__ scales,
       float* __restrict__ out) {
  const int e = blockIdx.z;
  const int cnt = counts[e];
  const int m0 = blockIdx.y * 64;
  if (m0 >= cnt) return;
  const int d0 = blockIdx.x * 64;

  __shared__ __bf16 lA[2][64 * LDS_STRIDE];
  __shared__ __bf16 lB[2][64 * LDS_STRIDE];

  const int tid = threadIdx.x;
  const int wv = tid >> 5, ln = tid & 31;
  const int mi = wv >> 2, ni = wv & 3;
  const int lr = ln & 15, lh = ln >> 4;

  const int srow = tid >> 2;
  const int skoff = (tid & 3) * 8;
  const int sofs = srow * LDS_STRIDE + skoff;

  int aslot = m0 + srow;
  if (aslot >= cnt) aslot = m0;
  const __bf16* aP = hbuf + ((size_t)e * TCAP + aslot) * FF + skoff;  // bf16 already
  const float*  bP = wd + ((size_t)e * DD + d0 + srow) * FF + skoff;

  v8f acc0 = {}, acc1 = {};

  {
    *(uint4*)(&lA[0][sofs]) = *(const uint4*)(aP);   // 8 bf16 = 16B straight copy
    float4 b0 = *(const float4*)(bP), b1 = *(const float4*)(bP + 4);
    *(v8bf*)(&lB[0][sofs]) = cvt8(b0, b1);
  }
  __syncthreads();

  int p = 0;
  for (int k0 = 0; k0 < FF; k0 += 32) {
    const bool more = (k0 + 32) < FF;
    uint4 nA;
    float4 nb0, nb1;
    if (more) {
      nA  = *(const uint4*)(aP + k0 + 32);
      nb0 = *(const float4*)(bP + k0 + 32);
      nb1 = *(const float4*)(bP + k0 + 36);
    }

    v16bf aF0 = load_a_frag(lA[p], mi * 32 + lr, lh);
    v16bf aF1 = load_a_frag(lA[p], mi * 32 + 16 + lr, lh);
    v16bf bF  = load_b_frag(lB[p], ni * 16 + lr, lh);

    acc0 = __builtin_amdgcn_wmma_f32_16x16x32_bf16(false, aF0, false, bF,
                                                   (short)0, acc0, false, false);
    acc1 = __builtin_amdgcn_wmma_f32_16x16x32_bf16(false, aF1, false, bF,
                                                   (short)0, acc1, false, false);

    if (more) {
      *(uint4*)(&lA[1 - p][sofs]) = nA;
      *(v8bf*)(&lB[1 - p][sofs])  = cvt8(nb0, nb1);
    }
    __syncthreads();
    p ^= 1;
  }

  const int col = d0 + ni * 16 + lr;
#pragma unroll
  for (int sub = 0; sub < 2; ++sub) {
    const v8f& a8 = sub ? acc1 : acc0;
#pragma unroll
    for (int r = 0; r < 8; ++r) {
      int slot = m0 + mi * 32 + sub * 16 + r + 8 * lh;
      if (slot < cnt) {
        int t = tokens[e * TCAP + slot];
        float s = scales[e * TCAP + slot];
        unsafeAtomicAdd(&out[(size_t)t * DD + col], s * a8[r]);
      }
    }
  }
}

// ---------------------------------------------------------------------------
// Launch
// ---------------------------------------------------------------------------
extern "C" void kernel_launch(void* const* d_in, const int* in_sizes, int n_in,
                              void* d_out, int out_size, void* d_ws, size_t ws_size,
                              hipStream_t stream) {
  const float* x  = (const float*)d_in[0];   // [B,S,D]
  const float* gw = (const float*)d_in[1];   // [E,D]
  const float* wg = (const float*)d_in[2];   // [E,F,D]
  const float* wu = (const float*)d_in[3];   // [E,F,D]
  const float* wd = (const float*)d_in[4];   // [E,D,F]

  float* out = (float*)d_out;                       // [T*D] then [T*E]
  float* logits = out + (size_t)TT * DD;

  // workspace carve (needs ~34 MB)
  char* ws = (char*)d_ws;
  int*    counts = (int*)(ws + 0);                       // 8 ints
  int*    tokens = (int*)(ws + 4096);                    // 8*2048 ints
  float*  scales = (float*)(ws + 4096 + NE * TCAP * 4);  // 8*2048 floats
  int*    sel    = (int*)(ws + 4096 + 2 * NE * TCAP * 4);                 // T*2 ints
  float*  selw   = (float*)(ws + 4096 + 2 * NE * TCAP * 4 + TT * 2 * 4);  // T*2 floats
  __bf16* hbuf   = (__bf16*)(ws + 262144);               // [E][TCAP][F] bf16 = 32 MB

  int nOut = TT * DD;
  k_zero<<<nOut / (256 * 4), 256, 0, stream>>>(out, nOut);
  k_router<<<TT / 8, 256, 0, stream>>>(x, gw, logits, sel, selw);
  k_build<<<NE, 32, 0, stream>>>(sel, selw, counts, tokens, scales);
  k_gateup<<<dim3(FF / 64, TT / 64, NE), 256, 0, stream>>>(x, wg, wu, counts, tokens, hbuf);
  k_down<<<dim3(DD / 64, TT / 64, NE), 256, 0, stream>>>(hbuf, wd, counts, tokens, scales, out);
}
